// GraphEncoder_2817498546216
// MI455X (gfx1250) — matrix operand order repaired and checked by reading, not compiled, not run on hardware
//
#include <hip/hip_runtime.h>
#include <hip/hip_bf16.h>

typedef __bf16 bf16_t;
typedef __attribute__((ext_vector_type(16))) __bf16 v16bf;
typedef __attribute__((ext_vector_type(8)))  __bf16 v8bf;
typedef __attribute__((ext_vector_type(4)))  __bf16 v4bf;
typedef __attribute__((ext_vector_type(8)))  float  v8f;
typedef __attribute__((ext_vector_type(4)))  float  v4f;
typedef __attribute__((ext_vector_type(4)))  unsigned int v4u;

// C[M x N] = act(A[M x K] (f32 row-major, cvt->bf16) @ Bcm[K x N] + bias)
//  - Bcm stored column-major ([N][K]); f32 (weights) or bf16 (activations).
//  - Optional outputs: f32 row-major, and/or bf16 transposed ([N][M]) feeding next pass.
//  - Workgroup: 16 rows x (NT*16) cols, NT waves. Grid.x = M/16 = 625.
//  - K-step 128, double-buffered LDS, register-prefetch of next A block, peeled tail,
//    all per-thread base addresses hoisted out of the K loop.
template<int NT, bool B_BF16>
__global__ __launch_bounds__(NT * 32)
void gemm_bf16_wmma(const float* __restrict__ A, int lda,
                    const void* __restrict__ Bv, int ldb,
                    const float* __restrict__ bias,
                    float* __restrict__ outF, int ldo,
                    bf16_t* __restrict__ outT, int ldt,
                    int K, int relu)
{
    constexpr int KSTEP  = 128;
    constexpr int LDSROW = KSTEP + 8;                 // 136 bf16 = 68 dwords: rows land on
                                                      // banks 4r..4r+3 -> conflict-free phases
    constexpr int CH = (16 * KSTEP / 4) / (NT * 32);  // 4-elem chunks per thread (2 or 4)
    __shared__ bf16_t Alds[2][16 * LDSROW];

    const int tid     = threadIdx.x;
    const int lane    = tid & 31;
    const int wave    = tid >> 5;
    const int halfsel = lane >> 4;       // 0: lanes 0-15, 1: lanes 16-31
    const int r       = lane & 15;
    const int mBase   = blockIdx.x * 16;
    const int nBase   = wave * 16;

    // ---- hoisted per-thread base pointers (loop-invariant) ----
    const float* aPtr[CH];       // A staging: global row base for each chunk (at k=0)
    bf16_t*      sPtr[CH];       // A staging: LDS destination (buffer 0)
    #pragma unroll
    for (int j = 0; j < CH; ++j) {
        const int c = tid + j * NT * 32;
        const int rr = c >> 5, kc = (c & 31) << 2;
        aPtr[j] = A + (size_t)(mBase + rr) * lda + kc;
        sPtr[j] = &Alds[0][rr * LDSROW + kc];
    }
    const bf16_t* bColH = (const bf16_t*)Bv + (size_t)(nBase + r) * ldb;
    const float*  bColF = (const float*)Bv  + (size_t)(nBase + r) * ldb;
    const int     aLdsBase = r * LDSROW + halfsel * 16;  // fragment base in LDS row
    constexpr int BUFELEMS = 16 * LDSROW;

    v8f acc = {0.f, 0.f, 0.f, 0.f, 0.f, 0.f, 0.f, 0.f};
    v4f pre[CH];

    const int nFull = K / KSTEP;
    const int kTail = K - nFull * KSTEP;

    // ---- next A block -> registers (streamed, non-temporal) ----
    auto preloadFull = [&](int k0) {
        #pragma unroll
        for (int j = 0; j < CH; ++j)
            pre[j] = __builtin_nontemporal_load((const v4f*)(aPtr[j] + k0));
    };
    auto preloadGuarded = [&](int k0) {
        #pragma unroll
        for (int j = 0; j < CH; ++j) {
            const int c = tid + j * NT * 32;
            const int kc = (c & 31) << 2;
            if (k0 + kc < K)      // K % 4 == 0 -> whole-chunk guard
                pre[j] = __builtin_nontemporal_load((const v4f*)(aPtr[j] + k0));
            else
                pre[j] = (v4f){0.f, 0.f, 0.f, 0.f};
        }
    };
    // ---- cvt + store prefetched block into LDS buffer ----
    auto stageStore = [&](int buf) {
        #pragma unroll
        for (int j = 0; j < CH; ++j) {
            v4bf p;
            #pragma unroll
            for (int i = 0; i < 4; ++i) p[i] = (bf16_t)pre[j][i];
            *(v4bf*)(sPtr[j] + buf * BUFELEMS) = p;
        }
    };
    // ---- 4 x WMMA over one 128-wide K block ----
    auto compute = [&](int buf, int k0, bool guarded) {
        const bf16_t* lbase = &Alds[buf][aLdsBase];
        #pragma unroll
        for (int h = 0; h < 4; ++h) {
            union { v4u u[2]; v16bf v; } af;
            af.u[0] = *(const v4u*)(lbase + h * 32);        // K +0..7  (+16..23)
            af.u[1] = *(const v4u*)(lbase + h * 32 + 8);    // K +8..15 (+24..31)

            union { v4u u[2]; v16bf v; } bbv;
            const int kb = k0 + h * 32 + halfsel * 16;      // 16 contiguous K per lane
            if (!guarded || kb < K) {                       // K % 16 == 0 -> whole guard
                if constexpr (B_BF16) {
                    bbv.u[0] = *(const v4u*)(bColH + kb);
                    bbv.u[1] = *(const v4u*)(bColH + kb + 8);
                } else {
                    const float* bp = bColF + kb;
                    v4f f0 = *(const v4f*)(bp);
                    v4f f1 = *(const v4f*)(bp + 4);
                    v4f f2 = *(const v4f*)(bp + 8);
                    v4f f3 = *(const v4f*)(bp + 12);
                    #pragma unroll
                    for (int i = 0; i < 4; ++i) {
                        bbv.v[i]      = (bf16_t)f0[i];
                        bbv.v[i + 4]  = (bf16_t)f1[i];
                        bbv.v[i + 8]  = (bf16_t)f2[i];
                        bbv.v[i + 12] = (bf16_t)f3[i];
                    }
                }
            } else {
                const v4u z = {0u, 0u, 0u, 0u};
                bbv.u[0] = z; bbv.u[1] = z;
            }
            acc = __builtin_amdgcn_wmma_f32_16x16x32_bf16(
                false, af.v, false, bbv.v, (short)0, acc, false, false);
        }
    };

    // ---- pipelined main loop (no guards) + peeled tail ----
    int buf = 0;
    if (nFull > 0) {
        preloadFull(0);
        stageStore(0);
        for (int i = 0; i < nFull; ++i) {
            __syncthreads();                       // buf staged; prev compute done
            const bool more = (i + 1 < nFull);
            if (more)        preloadFull((i + 1) * KSTEP);   // overlaps compute below
            else if (kTail)  preloadGuarded(nFull * KSTEP);
            compute(buf, i * KSTEP, false);
            if (more || kTail) stageStore(buf ^ 1);
            buf ^= 1;
        }
        if (kTail) {
            __syncthreads();
            compute(buf, nFull * KSTEP, true);
        }
    } else {                                       // K < 128 (projection head, K=64)
        preloadGuarded(0);
        stageStore(0);
        __syncthreads();
        compute(0, 0, true);
    }

    // ---- epilogue: bias, relu, stores ----
    const float bv = bias ? bias[nBase + r] : 0.f;
    #pragma unroll
    for (int v = 0; v < 8; ++v) {
        float c = acc[v] + bv;
        if (relu) c = fmaxf(c, 0.f);
        acc[v] = c;
    }
    const int mrow = mBase + halfsel * 8;          // C layout: row = v + 8*(lane>=16)
    if (outF) {
        #pragma unroll
        for (int v = 0; v < 8; ++v)
            outF[(size_t)(mrow + v) * ldo + nBase + r] = acc[v];
    }
    if (outT) {
        v8bf p;
        #pragma unroll
        for (int v = 0; v < 8; ++v) p[v] = (bf16_t)acc[v];
        *(v8bf*)&outT[(size_t)(nBase + r) * ldt + mrow] = p;   // 8 consecutive M -> b128
    }
}

extern "C" void kernel_launch(void* const* d_in, const int* in_sizes, int n_in,
                              void* d_out, int out_size, void* d_ws, size_t ws_size,
                              hipStream_t stream) {
    (void)in_sizes; (void)n_in; (void)out_size; (void)ws_size;

    const float* x   = (const float*)d_in[0];
    const float* Adj = (const float*)d_in[1];
    const float* W0  = (const float*)d_in[2];
    const float* b0  = (const float*)d_in[3];
    const float* W1  = (const float*)d_in[4];
    const float* b1  = (const float*)d_in[5];
    const float* W2  = (const float*)d_in[6];
    const float* b2  = (const float*)d_in[7];
    const float* Wp1 = (const float*)d_in[8];
    const float* bp1 = (const float*)d_in[9];
    const float* Wp2 = (const float*)d_in[10];
    const float* bp2 = (const float*)d_in[11];

    // workspace layout (bytes)
    char*   ws  = (char*)d_ws;
    bf16_t* T0t = (bf16_t*)(ws + 0);           // [128][10000] bf16   2.56 MB
    float*  H0  = (float*) (ws + 2560000);     // [10000][128] f32    5.12 MB
    bf16_t* T1t = (bf16_t*)(ws + 7680000);     // [128][10000] bf16   2.56 MB
    float*  H1  = (float*) (ws + 10240000);    // [10000][128] f32    5.12 MB
    bf16_t* T2t = (bf16_t*)(ws + 15360000);    // [64][10000]  bf16   1.28 MB
    float*  T3  = (float*) (ws + 16640000);    // [10000][64]  f32    2.56 MB

    float* z   = (float*)d_out;                // [10000][64] (tuple elem 0)
    float* emb = (float*)d_out + 640000;       // [10000][64] (tuple elem 1)

    const dim3 grid(625);  // 10000 / 16

    // 1) T0 = x @ W0^T + b0                 (K=512, N=128) -> bf16^T
    gemm_bf16_wmma<8, false><<<grid, 256, 0, stream>>>(x, 512,  W0, 512,  b0,
                                                       nullptr, 0, T0t, 10000, 512, 0);
    // 2) H0 = relu(Adj @ T0)                (K=10000, N=128)
    gemm_bf16_wmma<8, true ><<<grid, 256, 0, stream>>>(Adj, 10000, T0t, 10000, nullptr,
                                                       H0, 128, nullptr, 0, 10000, 1);
    // 3) T1 = H0 @ W1^T + b1                (K=128, N=128) -> bf16^T
    gemm_bf16_wmma<8, false><<<grid, 256, 0, stream>>>(H0, 128, W1, 128, b1,
                                                       nullptr, 0, T1t, 10000, 128, 0);
    // 4) H1 = relu(Adj @ T1)                (K=10000, N=128)
    gemm_bf16_wmma<8, true ><<<grid, 256, 0, stream>>>(Adj, 10000, T1t, 10000, nullptr,
                                                       H1, 128, nullptr, 0, 10000, 1);
    // 5) T2 = H1 @ W2^T + b2                (K=128, N=64) -> bf16^T
    gemm_bf16_wmma<4, false><<<grid, 128, 0, stream>>>(H1, 128, W2, 128, b2,
                                                       nullptr, 0, T2t, 10000, 128, 0);
    // 6) emb = Adj @ T2                     (K=10000, N=64, no relu) -> d_out
    gemm_bf16_wmma<4, true ><<<grid, 128, 0, stream>>>(Adj, 10000, T2t, 10000, nullptr,
                                                       emb, 64, nullptr, 0, 10000, 0);
    // 7) T3 = relu(emb @ Wp1^T + bp1)       (K=64, N=64)
    gemm_bf16_wmma<4, false><<<grid, 128, 0, stream>>>(emb, 64, Wp1, 64, bp1,
                                                       T3, 64, nullptr, 0, 64, 1);
    // 8) z = T3 @ Wp2^T + bp2               (K=64, N=64) -> d_out
    gemm_bf16_wmma<4, false><<<grid, 128, 0, stream>>>(T3, 64, Wp2, 64, bp2,
                                                       z, 64, nullptr, 0, 64, 0);
}
